// PointMamba_16106127360160
// MI455X (gfx1250) — matrix operand order repaired and checked by reading, not compiled, not run on hardware
//
#include <hip/hip_runtime.h>
#include <hip/hip_bf16.h>
#include <math.h>

// ---------------- model constants ----------------
#define BATCH 4
#define NPTS  8192
#define NG    512
#define KNB   32
#define FEATD 273
#define DM    384
#define NL    12
#define DSZ   16
#define DI    768
#define DTRK  24
#define NCLS  40
#define LTOT  (BATCH*NG)      // 2048 tokens
#define ENC_STRIDE 576        // padded K stride (multiple of 32, 16B-aligned rows)

typedef __attribute__((ext_vector_type(16))) _Float16 v16h;
typedef __attribute__((ext_vector_type(8)))  _Float16 v8h;
typedef __attribute__((ext_vector_type(8)))  float    v8f;

// ---------------- WMMA fragment helpers (CDNA5 16x16x32 f16 layouts) ----------------
// A (16x32 f16): lane m = lane&15; halves 0..7 -> K = koff..koff+7, halves 8..15 -> K = 16+koff..;
// koff = 8*(lane>=16). Two contiguous 16B runs per lane (LDS, f16, aligned).
__device__ __forceinline__ v16h load_a_frag(const _Float16* base, int stride, int m0, int kb, int lane) {
  int m = m0 + (lane & 15);
  int koff = (lane >> 4) << 3;
  const _Float16* p = base + (size_t)m * stride + kb + koff;
  v8h lo = *(const v8h*)p;
  v8h hi = *(const v8h*)(p + 16);
  return __builtin_shufflevector(lo, hi, 0,1,2,3,4,5,6,7,8,9,10,11,12,13,14,15);
}

// B (32x16 f16), B[k][n] = Wh[n][k], Wh pre-packed f16 [Npad][Kpad] (zero-padded):
// lane n = lane&15; halves 0..7 -> K = kb+koff..kb+koff+7, halves 8..15 -> K = kb+16+koff..; koff = 8*(lane>=16).
// Two 16B runs per lane -> two global b128 loads, no guards, no cvt.
__device__ __forceinline__ v16h load_b_frag_h(const _Float16* __restrict__ Wh, int ldwh,
                                              int n0, int kb, int lane) {
  const _Float16* p = Wh + (size_t)(n0 + (lane & 15)) * ldwh + kb + ((lane >> 4) << 3);
  v8h lo = *(const v8h*)p;
  v8h hi = *(const v8h*)(p + 16);
  return __builtin_shufflevector(lo, hi, 0,1,2,3,4,5,6,7,8,9,10,11,12,13,14,15);
}

// ordered-uint encoding so atomicMax(uint) == float max
__device__ __forceinline__ unsigned fenc(float f) {
  unsigned u = __float_as_uint(f);
  return (u & 0x80000000u) ? ~u : (u | 0x80000000u);
}
__device__ __forceinline__ float fdec(unsigned u) {
  unsigned v = (u & 0x80000000u) ? (u & 0x7FFFFFFFu) : ~u;
  return __uint_as_float(v);
}

__device__ __forceinline__ float siluf(float x) { return x / (1.0f + __expf(-x)); }

// ---------------- weight pre-pack: f32 [N][Kd] -> zero-padded f16 [Npad][Kpad] ----------------
__global__ void pack_w_kernel(const float* __restrict__ W, _Float16* __restrict__ Wh,
                              int N, int Kd, int Npad, int Kpad) {
  int i = blockIdx.x * blockDim.x + threadIdx.x;
  if (i >= Npad * Kpad) return;
  int o = i / Kpad, k = i % Kpad;
  Wh[i] = (o < N && k < Kd) ? (_Float16)W[(size_t)o * Kd + k] : (_Float16)0.0f;
}

// ---------------- FPS: one block per batch, sequential 512 picks ----------------
__global__ __launch_bounds__(1024) void fps_kernel(const float* __restrict__ xyz,
                                                   int* __restrict__ fps_idx,
                                                   float* __restrict__ centers) {
  int b = blockIdx.x, t = threadIdx.x;
  __shared__ float rv[1024];
  __shared__ int   ri[1024];
  const float* X = xyz + (size_t)b * NPTS * 3;
  float px[8], py[8], pz[8], ld[8];
#pragma unroll
  for (int j = 0; j < 8; ++j) {
    int i = j * 1024 + t;
    px[j] = X[i*3+0]; py[j] = X[i*3+1]; pz[j] = X[i*3+2];
    ld[j] = 1e10f;
  }
  int last = 0;
  for (int it = 0; it < NG; ++it) {
    float cx = X[last*3+0], cy = X[last*3+1], cz = X[last*3+2];
    if (t == 0) {
      fps_idx[b*NG + it] = last;
      centers[(size_t)(b*NG + it)*3 + 0] = cx;
      centers[(size_t)(b*NG + it)*3 + 1] = cy;
      centers[(size_t)(b*NG + it)*3 + 2] = cz;
    }
    float bv = -1.0f; int bi = 0;
#pragma unroll
    for (int j = 0; j < 8; ++j) {
      float dx = px[j]-cx, dy = py[j]-cy, dz = pz[j]-cz;
      float d = dx*dx + dy*dy + dz*dz;
      ld[j] = fminf(ld[j], d);
      int i = j * 1024 + t;
      if (ld[j] > bv || (ld[j] == bv && i < bi)) { bv = ld[j]; bi = i; }
    }
    rv[t] = bv; ri[t] = bi;
    __syncthreads();
    for (int s = 512; s > 0; s >>= 1) {
      if (t < s) {
        float ov = rv[t+s]; int oi = ri[t+s];
        if (ov > rv[t] || (ov == rv[t] && oi < ri[t])) { rv[t] = ov; ri[t] = oi; }
      }
      __syncthreads();
    }
    last = ri[0];
    __syncthreads();
  }
}

// ---------------- KNN: one block per (b,g); distances in LDS; 32 argmin passes ----------------
__global__ __launch_bounds__(256) void knn_kernel(const float* __restrict__ xyz,
                                                  const float* __restrict__ centers,
                                                  int* __restrict__ nidx) {
  __shared__ float sd[NPTS];
  __shared__ float rv[256];
  __shared__ int   ri[256];
  int bg = blockIdx.x, t = threadIdx.x;
  int b = bg >> 9;
  const float* X = xyz + (size_t)b * NPTS * 3;
  float cx = centers[(size_t)bg*3+0], cy = centers[(size_t)bg*3+1], cz = centers[(size_t)bg*3+2];
  for (int i = t; i < NPTS; i += 256) {
    float dx = X[i*3+0]-cx, dy = X[i*3+1]-cy, dz = X[i*3+2]-cz;
    sd[i] = dx*dx + dy*dy + dz*dz;
  }
  __syncthreads();
  for (int kk = 0; kk < KNB; ++kk) {
    float bv = 3.4e38f; int bi = 0;
    for (int i = t; i < NPTS; i += 256) {
      float v = sd[i];
      if (v < bv || (v == bv && i < bi)) { bv = v; bi = i; }
    }
    rv[t] = bv; ri[t] = bi;
    __syncthreads();
    for (int s = 128; s > 0; s >>= 1) {
      if (t < s) {
        if (rv[t+s] < rv[t] || (rv[t+s] == rv[t] && ri[t+s] < ri[t])) { rv[t] = rv[t+s]; ri[t] = ri[t+s]; }
      }
      __syncthreads();
    }
    if (t == 0) { nidx[(size_t)bg*KNB + kk] = ri[0]; sd[ri[0]] = 3.4e38f; }
    __syncthreads();
  }
}

// ---------------- fused gather + encoder: one block per group, 8 waves, 2 WMMA GEMMs in LDS ----------------
// Each wave owns an n-tile and accumulates BOTH m-tiles against one B fragment (halves weight traffic).
__global__ __launch_bounds__(256) void encoder_kernel(const float* __restrict__ feats,
                                                      const int* __restrict__ nidx,
                                                      const _Float16* __restrict__ w1h, const float* __restrict__ b1,
                                                      const float* __restrict__ bng, const float* __restrict__ bnb,
                                                      const _Float16* __restrict__ w2h, const float* __restrict__ b2,
                                                      float* __restrict__ tokens) {
  __shared__ __align__(16) unsigned char smem[2*32*ENC_STRIDE*2 + DM*4];  // 75264 B
  _Float16* sH    = (_Float16*)smem;                         // [32][576] layer1 out (f16)
  _Float16* sFeat = (_Float16*)(smem + 32*ENC_STRIDE*2);     // [32][576] concat features (f16)
  unsigned* sTok  = (unsigned*)(smem + 2*32*ENC_STRIDE*2);   // [384] ordered-uint max
  int bg = blockIdx.x;
  int b  = bg >> 9;
  int t = threadIdx.x, lane = t & 31, w = t >> 5;

  // gather x into cols 273..545
  for (int idx = t; idx < 32 * FEATD; idx += 256) {
    int r = idx / FEATD, c = idx % FEATD;
    int pt = nidx[(size_t)bg*KNB + r];
    sFeat[r*ENC_STRIDE + FEATD + c] = (_Float16)feats[((size_t)b*NPTS + pt)*FEATD + c];
  }
  // zero K-pad cols 546..575 of both buffers, init token max
  for (int idx = t; idx < 32 * 30; idx += 256) {
    int r = idx / 30, c = 546 + idx % 30;
    sFeat[r*ENC_STRIDE + c] = (_Float16)0.0f;
    sH[r*ENC_STRIDE + c]    = (_Float16)0.0f;
  }
  for (int n = t; n < DM; n += 256) sTok[n] = 0u;
  __syncthreads();

  // group max -> broadcast into cols 0..272
  for (int c = t; c < FEATD; c += 256) {
    float mx = -3.4e38f;
    for (int r = 0; r < 32; ++r) mx = fmaxf(mx, (float)sFeat[r*ENC_STRIDE + FEATD + c]);
    _Float16 hm = (_Float16)mx;
    for (int r = 0; r < 32; ++r) sFeat[r*ENC_STRIDE + c] = hm;
  }
  __syncthreads();

  const float bnrs = rsqrtf(1.0f + 1e-5f);
  // layer1: h[32,546] = relu(bn(feat[32,546] @ w1^T + b1)) -> sH (f16); w1h is [560][576] f16
  for (int nt = w; nt < 35; nt += 8) {
    v8f acc0 = {}, acc1 = {};
    for (int kb = 0; kb < ENC_STRIDE; kb += 32) {
      v16h bf  = load_b_frag_h(w1h, ENC_STRIDE, nt*16, kb, lane);
      v16h af0 = load_a_frag(sFeat, ENC_STRIDE, 0,  kb, lane);
      v16h af1 = load_a_frag(sFeat, ENC_STRIDE, 16, kb, lane);
      acc0 = __builtin_amdgcn_wmma_f32_16x16x32_f16(false, af0, false, bf, (short)0, acc0, false, false);
      acc1 = __builtin_amdgcn_wmma_f32_16x16x32_f16(false, af1, false, bf, (short)0, acc1, false, false);
    }
    int n = nt*16 + (lane & 15);
    if (n < 546) {
      float scale = bng[n] * bnrs, bias1 = b1[n], shift = bnb[n];
#pragma unroll
      for (int r = 0; r < 8; ++r) {
        int m = r + ((lane >> 4) << 3);
        float v0 = fmaxf((acc0[r] + bias1) * scale + shift, 0.0f);
        float v1 = fmaxf((acc1[r] + bias1) * scale + shift, 0.0f);
        sH[m*ENC_STRIDE + n]        = (_Float16)v0;
        sH[(m+16)*ENC_STRIDE + n]   = (_Float16)v1;
      }
    }
  }
  __syncthreads();

  // layer2: out[32,384] = h @ w2^T + b2 ; token = max over rows; w2h is [384][576] f16
  for (int nt = w; nt < 24; nt += 8) {
    v8f acc0 = {}, acc1 = {};
    for (int kb = 0; kb < ENC_STRIDE; kb += 32) {
      v16h bf  = load_b_frag_h(w2h, ENC_STRIDE, nt*16, kb, lane);
      v16h af0 = load_a_frag(sH, ENC_STRIDE, 0,  kb, lane);
      v16h af1 = load_a_frag(sH, ENC_STRIDE, 16, kb, lane);
      acc0 = __builtin_amdgcn_wmma_f32_16x16x32_f16(false, af0, false, bf, (short)0, acc0, false, false);
      acc1 = __builtin_amdgcn_wmma_f32_16x16x32_f16(false, af1, false, bf, (short)0, acc1, false, false);
    }
    int n = nt*16 + (lane & 15);
    float bias2 = b2[n];
    float mx = -3.4e38f;
#pragma unroll
    for (int r = 0; r < 8; ++r) {
      mx = fmaxf(mx, fmaxf(acc0[r] + bias2, acc1[r] + bias2));
    }
    atomicMax(&sTok[n], fenc(mx));
  }
  __syncthreads();
  for (int n = t; n < DM; n += 256) tokens[(size_t)bg*DM + n] = fdec(sTok[n]);
}

// ---------------- generic WMMA GEMM: C[M,N] = act(A[M,Kd] @ W[N,Kd]^T + bias) ----------------
// Wh: pre-packed f16 [Npad][Kpad], Npad >= gridDim.x*128, Kpad = roundup(Kd,32), zero-padded.
// block = 8 waves, 64x128 output tile: 4 m-tiles per wave against one B fragment per k-chunk
// (4x weight-traffic reduction, 4 independent WMMA chains per wave).
__global__ __launch_bounds__(256) void wmma_gemm_kernel(const float* __restrict__ A, int lda,
                                                        const _Float16* __restrict__ Wh, int Kpad,
                                                        const float* __restrict__ bias,
                                                        float* __restrict__ C, int ldc,
                                                        int M, int N, int Kd, int act) {
  __shared__ __align__(16) _Float16 sA[64*32];
  int t = threadIdx.x, lane = t & 31, w = t >> 5;
  int m0 = blockIdx.y * 64;
  int n0 = blockIdx.x * 128 + w * 16;
  v8f acc[4] = {};
  for (int kb = 0; kb < Kpad; kb += 32) {
    // stage 64x32 A chunk as f16 (adjacent half-pairs -> mergeable ds_store_b32)
#pragma unroll
    for (int j = 0; j < 4; ++j) {
      int p = t + j * 256;             // pair index, 1024 pairs
      int r = p >> 4, c = (p & 15) * 2;
      int m = m0 + r, k = kb + c;
      _Float16 h0 = (m < M && k     < Kd) ? (_Float16)A[(size_t)m*lda + k]     : (_Float16)0.0f;
      _Float16 h1 = (m < M && (k+1) < Kd) ? (_Float16)A[(size_t)m*lda + k + 1] : (_Float16)0.0f;
      sA[r*32 + c]     = h0;
      sA[r*32 + c + 1] = h1;
    }
    __syncthreads();
    v16h bf = load_b_frag_h(Wh, Kpad, n0, kb, lane);
#pragma unroll
    for (int mt = 0; mt < 4; ++mt) {
      v16h af = load_a_frag(sA, 32, mt*16, 0, lane);
      acc[mt] = __builtin_amdgcn_wmma_f32_16x16x32_f16(false, af, false, bf, (short)0, acc[mt], false, false);
    }
    __syncthreads();
  }
  int n = n0 + (lane & 15);
#pragma unroll
  for (int mt = 0; mt < 4; ++mt) {
#pragma unroll
    for (int r = 0; r < 8; ++r) {
      int m = m0 + mt*16 + r + ((lane >> 4) << 3);
      if (m < M && n < N) {
        float v = acc[mt][r] + (bias ? bias[n] : 0.0f);
        if (act == 1)      v = siluf(v);
        else if (act == 2) v = (v > 20.0f) ? v : log1pf(__expf(v));   // softplus
        C[(size_t)m*ldc + n] = v;
      }
    }
  }
}

// ---------------- residual add + LayerNorm (row per block) ----------------
__global__ __launch_bounds__(128) void add_ln_kernel(float* __restrict__ residual,
                                                     const float* __restrict__ hidden,
                                                     const float* __restrict__ g,
                                                     const float* __restrict__ bta,
                                                     float* __restrict__ out) {
  int row = blockIdx.x, t = threadIdx.x;
  __shared__ float s1[128], s2[128];
  float* rrow = residual + (size_t)row * DM;
  const float* hrow = hidden + (size_t)row * DM;
  float vals[3], sum = 0.0f, sq = 0.0f;
#pragma unroll
  for (int j = 0; j < 3; ++j) {
    int c = t + j * 128;
    float v = rrow[c] + hrow[c];
    rrow[c] = v;
    vals[j] = v;
    sum += v; sq += v * v;
  }
  s1[t] = sum; s2[t] = sq;
  __syncthreads();
  for (int s = 64; s > 0; s >>= 1) {
    if (t < s) { s1[t] += s1[t+s]; s2[t] += s2[t+s]; }
    __syncthreads();
  }
  float mean = s1[0] * (1.0f / DM);
  float var  = s2[0] * (1.0f / DM) - mean * mean;
  float inv  = rsqrtf(var + 1e-5f);
#pragma unroll
  for (int j = 0; j < 3; ++j) {
    int c = t + j * 128;
    out[(size_t)row*DM + c] = (vals[j] - mean) * inv * g[c] + bta[c];
  }
}

// ---------------- causal depthwise conv1d (k=4) + silu ----------------
__global__ void conv_silu_kernel(const float* __restrict__ xz,
                                 const float* __restrict__ cw,
                                 const float* __restrict__ cb,
                                 float* __restrict__ xc) {
  int i = blockIdx.x * blockDim.x + threadIdx.x;
  if (i >= LTOT * DI) return;
  int d = i % DI, row = i / DI;
  int l = row & (NG - 1), b = row >> 9;
  float acc = cb[d];
#pragma unroll
  for (int k = 0; k < 4; ++k) {
    int ls = l - 3 + k;
    if (ls >= 0) acc += xz[(size_t)(b*NG + ls)*(2*DI) + d] * cw[d*4 + k];
  }
  xc[i] = siluf(acc);
}

// ---------------- selective scan: thread owns one (b,d); h[16],A[16] in regs; B/C via LDS ----------------
__global__ __launch_bounds__(256) void scan_kernel(const float* __restrict__ dt,
                                                   const float* __restrict__ dbc,
                                                   const float* __restrict__ xc,
                                                   const float* __restrict__ A_log,
                                                   const float* __restrict__ Dp,
                                                   float* __restrict__ y) {
  int blk = blockIdx.x;                 // BATCH * 3
  int b = blk / 3, dch = blk % 3;
  int t = threadIdx.x;
  int d = dch * 256 + t;
  float a[DSZ];
#pragma unroll
  for (int s = 0; s < DSZ; ++s) a[s] = -__expf(A_log[(size_t)d*DSZ + s]);
  float Dd = Dp[d];
  float h[DSZ];
#pragma unroll
  for (int s = 0; s < DSZ; ++s) h[s] = 0.0f;
  __shared__ float sB[DSZ], sC[DSZ];
  for (int l = 0; l < NG; ++l) {
    size_t row = (size_t)b*NG + l;
    if (t < DSZ)            sB[t]        = dbc[row*56 + DTRK + t];
    else if (t < 2*DSZ)     sC[t - DSZ]  = dbc[row*56 + DTRK + DSZ + (t - DSZ)];
    __syncthreads();
    float dtv = dt[row*DI + d];
    float u   = xc[row*DI + d];
    float acc = 0.0f;
#pragma unroll
    for (int s = 0; s < DSZ; ++s) {
      h[s] = __expf(dtv * a[s]) * h[s] + dtv * sB[s] * u;
      acc += h[s] * sC[s];
    }
    y[row*DI + d] = acc + u * Dd;
    __syncthreads();
  }
}

// ---------------- gate: yg = y * silu(z) ----------------
__global__ void gate_kernel(const float* __restrict__ y,
                            const float* __restrict__ xz,
                            float* __restrict__ yg) {
  int i = blockIdx.x * blockDim.x + threadIdx.x;
  if (i >= LTOT * DI) return;
  int d = i % DI, row = i / DI;
  float z = xz[(size_t)row*(2*DI) + DI + d];
  yg[i] = y[i] * siluf(z);
}

__global__ void zero_kernel(float* __restrict__ p, int n) {
  int i = blockIdx.x * blockDim.x + threadIdx.x;
  if (i < n) p[i] = 0.0f;
}

// ---------------- final pooling + head ----------------
__global__ __launch_bounds__(DM) void pool_kernel(const float* __restrict__ hfin,
                                                  float* __restrict__ pooled) {
  int b = blockIdx.x, c = threadIdx.x;
  float mx = -3.4e38f, sm = 0.0f;
  for (int l = 0; l < NG; ++l) {
    float v = hfin[((size_t)b*NG + l)*DM + c];
    mx = fmaxf(mx, v); sm += v;
  }
  pooled[(size_t)b*(2*DM) + c]       = mx;
  pooled[(size_t)b*(2*DM) + DM + c]  = sm * (1.0f / NG);
}

__global__ void head_kernel(const float* __restrict__ pooled,
                            const float* __restrict__ hw,
                            const float* __restrict__ hb,
                            float* __restrict__ out) {
  int t = threadIdx.x;
  if (t >= BATCH * NCLS) return;
  int b = t / NCLS, o = t % NCLS;
  float acc = hb[o];
  for (int k = 0; k < 2*DM; ++k) acc += pooled[(size_t)b*(2*DM) + k] * hw[(size_t)o*(2*DM) + k];
  out[b*NCLS + o] = acc;
}

// ---------------- launcher ----------------
extern "C" void kernel_launch(void* const* d_in, const int* in_sizes, int n_in,
                              void* d_out, int out_size, void* d_ws, size_t ws_size,
                              hipStream_t stream) {
  const float* xyz     = (const float*)d_in[0];
  const float* feats   = (const float*)d_in[1];
  const float* enc_w1  = (const float*)d_in[2];
  const float* enc_b1  = (const float*)d_in[3];
  const float* enc_bng = (const float*)d_in[4];
  const float* enc_bnb = (const float*)d_in[5];
  const float* enc_w2  = (const float*)d_in[6];
  const float* enc_b2  = (const float*)d_in[7];
  const float* normf_g = (const float*)d_in[8 + 11*NL + 0];
  const float* normf_b = (const float*)d_in[8 + 11*NL + 1];
  const float* head_w  = (const float*)d_in[8 + 11*NL + 2];
  const float* head_b  = (const float*)d_in[8 + 11*NL + 3];
  float* out = (float*)d_out;

  char* base = (char*)d_ws;
  size_t off = 0;
  auto alloc = [&](size_t bytes) -> void* {
    void* p = base + off;
    off = (off + bytes + 255) & ~(size_t)255;
    return p;
  };
  int*   fps_idx  = (int*)  alloc((size_t)LTOT * 4);
  float* centers  = (float*)alloc((size_t)LTOT * 3 * 4);
  int*   nidx     = (int*)  alloc((size_t)LTOT * KNB * 4);
  float* hidden   = (float*)alloc((size_t)LTOT * DM * 4);
  float* residual = (float*)alloc((size_t)LTOT * DM * 4);
  float* xln      = (float*)alloc((size_t)LTOT * DM * 4);
  float* xz       = (float*)alloc((size_t)LTOT * 2 * DI * 4);
  float* xc       = (float*)alloc((size_t)LTOT * DI * 4);
  float* dbc      = (float*)alloc((size_t)LTOT * 56 * 4);
  float* dtb      = (float*)alloc((size_t)LTOT * DI * 4);
  float* ybuf     = (float*)alloc((size_t)LTOT * DI * 4);
  float* yg       = (float*)alloc((size_t)LTOT * DI * 4);
  float* hfin     = (float*)alloc((size_t)LTOT * DM * 4);
  float* pooled   = (float*)alloc((size_t)BATCH * 2 * DM * 4);

  // packed f16 weights (zero-padded)
  _Float16* w1h = (_Float16*)alloc((size_t)560 * ENC_STRIDE * 2);   // enc_w1 [546,546] -> [560][576]
  _Float16* w2h = (_Float16*)alloc((size_t)DM * ENC_STRIDE * 2);    // enc_w2 [384,546] -> [384][576]
  _Float16* inph[NL]; _Float16* xph[NL]; _Float16* dtwh[NL]; _Float16* outph[NL];
  for (int l = 0; l < NL; ++l) {
    inph[l]  = (_Float16*)alloc((size_t)(2*DI) * DM * 2);   // [1536][384]
    xph[l]   = (_Float16*)alloc((size_t)128 * DI * 2);      // [56,768] -> [128][768]
    dtwh[l]  = (_Float16*)alloc((size_t)DI * 32 * 2);       // [768,24] -> [768][32]
    outph[l] = (_Float16*)alloc((size_t)DM * DI * 2);       // [384][768]
  }

  // ---- pre-pack all weights (trivial bandwidth; runs every launch, deterministic) ----
  {
    int n;
    n = 560 * ENC_STRIDE;
    pack_w_kernel<<<(n+255)/256, 256, 0, stream>>>(enc_w1, w1h, 546, 546, 560, ENC_STRIDE);
    n = DM * ENC_STRIDE;
    pack_w_kernel<<<(n+255)/256, 256, 0, stream>>>(enc_w2, w2h, DM, 546, DM, ENC_STRIDE);
    for (int l = 0; l < NL; ++l) {
      const float* in_proj  = (const float*)d_in[8 + 11*l + 2];
      const float* x_proj   = (const float*)d_in[8 + 11*l + 5];
      const float* dt_w     = (const float*)d_in[8 + 11*l + 6];
      const float* out_proj = (const float*)d_in[8 + 11*l + 10];
      n = (2*DI) * DM;
      pack_w_kernel<<<(n+255)/256, 256, 0, stream>>>(in_proj, inph[l], 2*DI, DM, 2*DI, DM);
      n = 128 * DI;
      pack_w_kernel<<<(n+255)/256, 256, 0, stream>>>(x_proj, xph[l], 56, DI, 128, DI);
      n = DI * 32;
      pack_w_kernel<<<(n+255)/256, 256, 0, stream>>>(dt_w, dtwh[l], DI, DTRK, DI, 32);
      n = DM * DI;
      pack_w_kernel<<<(n+255)/256, 256, 0, stream>>>(out_proj, outph[l], DM, DI, DM, DI);
    }
  }

  fps_kernel<<<BATCH, 1024, 0, stream>>>(xyz, fps_idx, centers);
  knn_kernel<<<LTOT, 256, 0, stream>>>(xyz, centers, nidx);
  encoder_kernel<<<LTOT, 256, 0, stream>>>(feats, nidx, w1h, enc_b1, enc_bng, enc_bnb,
                                           w2h, enc_b2, hidden);
  {
    int n = LTOT * DM;
    zero_kernel<<<(n + 255) / 256, 256, 0, stream>>>(residual, n);
  }
  for (int l = 0; l < NL; ++l) {
    const float* norm_g   = (const float*)d_in[8 + 11*l + 0];
    const float* norm_b   = (const float*)d_in[8 + 11*l + 1];
    const float* conv_w   = (const float*)d_in[8 + 11*l + 3];
    const float* conv_b   = (const float*)d_in[8 + 11*l + 4];
    const float* dt_b     = (const float*)d_in[8 + 11*l + 7];
    const float* A_log    = (const float*)d_in[8 + 11*l + 8];
    const float* Dp       = (const float*)d_in[8 + 11*l + 9];

    add_ln_kernel<<<LTOT, 128, 0, stream>>>(residual, hidden, norm_g, norm_b, xln);
    // xz[2048,1536] = xln @ in_proj^T        (Kpad = 384), 64x128 tiles
    wmma_gemm_kernel<<<dim3(12, 32), 256, 0, stream>>>(xln, DM, inph[l], DM, nullptr,
                                                       xz, 2*DI, LTOT, 2*DI, DM, 0);
    {
      int n = LTOT * DI;
      conv_silu_kernel<<<(n + 255) / 256, 256, 0, stream>>>(xz, conv_w, conv_b, xc);
    }
    // dbc[2048,56] = xc @ x_proj^T           (Kpad = 768, Npad = 128)
    wmma_gemm_kernel<<<dim3(1, 32), 256, 0, stream>>>(xc, DI, xph[l], DI, nullptr,
                                                      dbc, 56, LTOT, 56, DI, 0);
    // dt[2048,768] = softplus(dbc[:, :24] @ dt_w^T + dt_b)   (Kpad = 32)
    wmma_gemm_kernel<<<dim3(6, 32), 256, 0, stream>>>(dbc, 56, dtwh[l], 32, dt_b,
                                                      dtb, DI, LTOT, DI, DTRK, 2);
    scan_kernel<<<BATCH * 3, 256, 0, stream>>>(dtb, dbc, xc, A_log, Dp, ybuf);
    {
      int n = LTOT * DI;
      gate_kernel<<<(n + 255) / 256, 256, 0, stream>>>(ybuf, xz, yg);
    }
    // hidden[2048,384] = yg @ out_proj^T     (Kpad = 768)
    wmma_gemm_kernel<<<dim3(3, 32), 256, 0, stream>>>(yg, DI, outph[l], DI, nullptr,
                                                      hidden, DM, LTOT, DM, DI, 0);
  }
  add_ln_kernel<<<LTOT, 128, 0, stream>>>(residual, hidden, normf_g, normf_b, hfin);
  pool_kernel<<<BATCH, DM, 0, stream>>>(hfin, pooled);
  head_kernel<<<1, 192, 0, stream>>>(pooled, head_w, head_b, out);
  (void)in_sizes; (void)n_in; (void)out_size; (void)ws_size;
}
